// AttentionLayer_4509715661568
// MI455X (gfx1250) — matrix operand order repaired and checked by reading, not compiled, not run on hardware
//
#include <hip/hip_runtime.h>
#include <hip/hip_bf16.h>

#define BATCH  4
#define SEQ    2048
#define NH     12
#define HD     64
#define DMODEL 768

typedef __bf16 bf16;
typedef __attribute__((ext_vector_type(16))) __bf16 v16bf;
typedef __attribute__((ext_vector_type(8)))  __bf16 v8bf;
typedef __attribute__((ext_vector_type(8)))  float  v8f;
typedef __attribute__((ext_vector_type(4)))  float  v4f;
typedef unsigned int v4u __attribute__((ext_vector_type(4)));
typedef int          v4i __attribute__((ext_vector_type(4)));
typedef int          v8i __attribute__((ext_vector_type(8)));

#if defined(__has_builtin)
#if __has_builtin(__builtin_amdgcn_tensor_load_to_lds)
#define HAVE_TDM 1
#endif
#endif

#define WMMA_BF16(a, b, c) \
  __builtin_amdgcn_wmma_f32_16x16x32_bf16(false, (a), false, (b), (short)0, (c), false, false)

__device__ __forceinline__ bf16 f2bf(float f) {
  unsigned u = __builtin_bit_cast(unsigned, f);
  unsigned r = u + 0x7FFFu + ((u >> 16) & 1u);   // round-to-nearest-even
  unsigned short h = (unsigned short)(r >> 16);
  return __builtin_bit_cast(bf16, h);
}

__device__ __forceinline__ v8f zero8() {
  v8f z = {0.f, 0.f, 0.f, 0.f, 0.f, 0.f, 0.f, 0.f};
  return z;
}

// A-fragment (16x32 bf16, row-major source, k contiguous):
// lane: row = lane&15; VGPR0-7 hold k = kb..kb+15, kb=(lane>>4)*16
__device__ __forceinline__ v16bf load_a32(const bf16* p, int ld, int lane) {
  const bf16* row = p + (size_t)(lane & 15) * ld + ((lane >> 4) * 16);
  v8bf lo = *(const v8bf*)row;
  v8bf hi = *(const v8bf*)(row + 8);
  v16bf out;
#pragma unroll
  for (int i = 0; i < 8; ++i) { out[i] = lo[i]; out[i + 8] = hi[i]; }
  return out;
}

// B-fragment (32x16 bf16) from Bt[n][k] (k contiguous):
// lane: col n = lane&15; holds 16 consecutive k at kb=(lane>>4)*16  -> one 32B load
__device__ __forceinline__ v16bf load_b32(const bf16* p, int ld, int lane) {
  return *(const v16bf*)(p + (size_t)(lane & 15) * ld + ((lane >> 4) * 16));
}

#ifdef HAVE_TDM
// ---------------------------------------------------------------------------
// Tensor Data Mover: 2D tile (dim0 contiguous run, dim1 rows with stride0),
// bf16 elements, packed into LDS at lds_off.  D# per CDNA5 ISA.
// ---------------------------------------------------------------------------
__device__ __forceinline__ void tdm_load_2d(unsigned lds_off, const void* gaddr,
                                            unsigned tensor_d0, unsigned tensor_d1,
                                            unsigned tile_d0, unsigned tile_d1,
                                            unsigned long long stride0) {
  unsigned long long ga = (unsigned long long)(size_t)gaddr;
  // group0: count=1 (valid), is_restore=0, gather off; lds_addr; 57b global addr; type=2
  v4u g0 = { 0x1u,
             lds_off,
             (unsigned)(ga & 0xFFFFFFFFu),
             (unsigned)(((ga >> 32) & 0x01FFFFFFu) | (2u << 30)) };
  // group1: wg_mask=0, data_size=1 (2B), no barrier/iterate/pad
  unsigned long long s1 = stride0;   // dim1 stride (unused for 2D tiles)
  v8i g1 = { (int)(1u << 16),
             (int)((tensor_d0 & 0xFFFFu) << 16),
             (int)((tensor_d0 >> 16) | ((tensor_d1 & 0xFFFFu) << 16)),
             (int)((tensor_d1 >> 16) | ((tile_d0 & 0xFFFFu) << 16)),
             (int)(tile_d1 & 0xFFFFu),                       // tile_dim2 = 0
             (int)(stride0 & 0xFFFFFFFFu),
             (int)((unsigned)(stride0 >> 32) | ((unsigned)(s1 & 0xFFFFu) << 16)),
             (int)((unsigned)(s1 >> 16)) };
  v4i zz = {0, 0, 0, 0};
#if __clang_major__ >= 23
  v8i zz8 = {0, 0, 0, 0, 0, 0, 0, 0};
  __builtin_amdgcn_tensor_load_to_lds(g0, g1, zz, zz, zz8, 0);
#else
  __builtin_amdgcn_tensor_load_to_lds(g0, g1, zz, zz, 0);
#endif
}
#endif

// ---------------------------------------------------------------------------
// Kernel 1: fused QKV projection.  Y = x @ W + b, f32 -> bf16 workspace.
// z = 0/1 -> Q/K stored [b,h,s,d];  z = 2 -> V stored transposed [b,h,d,s].
// Workgroup: 128 threads (4 waves), 128(M) x 64(N) tile, K-step 32.
// ---------------------------------------------------------------------------
__global__ __launch_bounds__(128) void qkv_proj_kernel(
    const float* __restrict__ x,
    const float* __restrict__ Wq, const float* __restrict__ bq,
    const float* __restrict__ Wk, const float* __restrict__ bk,
    const float* __restrict__ Wv, const float* __restrict__ bv,
    bf16* __restrict__ qws, bf16* __restrict__ kws, bf16* __restrict__ vws) {
  __shared__ __align__(16) bf16 As[128][32];   // x tile, row-major
  __shared__ __align__(16) bf16 Bs[64][32];    // W tile, transposed: Bs[n][k]

  const int t    = threadIdx.x;
  const int lane = t & 31;
  const int wave = t >> 5;
  const int m0   = blockIdx.x * 128;
  const int n0   = blockIdx.y * 64;
  const int z    = blockIdx.z;

  const float* W    = (z == 0) ? Wq : (z == 1) ? Wk : Wv;
  const float* bias = (z == 0) ? bq : (z == 1) ? bk : bv;

  v8f c[2][4];
#pragma unroll
  for (int i = 0; i < 2; ++i)
#pragma unroll
    for (int j = 0; j < 4; ++j) c[i][j] = zero8();

  const int bn  = t & 63;
  const int bk0 = (t >> 6) * 16;

  for (int k0 = 0; k0 < DMODEL; k0 += 32) {
    // stage x tile: thread t owns row t (32 f32 -> 32 bf16)
    const float* srcA = x + (size_t)(m0 + t) * DMODEL + k0;
    if (k0 + 32 < DMODEL) __builtin_prefetch(srcA + 32, 0, 0);  // global_prefetch_b8
#pragma unroll
    for (int i = 0; i < 32; i += 4) {
      v4f v = *(const v4f*)(srcA + i);
      As[t][i + 0] = f2bf(v[0]); As[t][i + 1] = f2bf(v[1]);
      As[t][i + 2] = f2bf(v[2]); As[t][i + 3] = f2bf(v[3]);
    }
    // stage W tile transposed: Bs[n][k] = W[k0+k][n0+n]
#pragma unroll
    for (int i = 0; i < 16; ++i)
      Bs[bn][bk0 + i] = f2bf(W[(size_t)(k0 + bk0 + i) * DMODEL + (n0 + bn)]);
    __syncthreads();

    v16bf a0 = load_a32(&As[wave * 32][0],      32, lane);
    v16bf a1 = load_a32(&As[wave * 32 + 16][0], 32, lane);
#pragma unroll
    for (int nt = 0; nt < 4; ++nt) {
      v16bf bb = load_b32(&Bs[nt * 16][0], 32, lane);
      c[0][nt] = WMMA_BF16(a0, bb, c[0][nt]);
      c[1][nt] = WMMA_BF16(a1, bb, c[1][nt]);
    }
    __syncthreads();
  }

  // epilogue: bias add + scatter into attention-friendly layouts
#pragma unroll
  for (int mt = 0; mt < 2; ++mt) {
#pragma unroll
    for (int nt = 0; nt < 4; ++nt) {
      const int   gn = n0 + nt * 16 + (lane & 15);
      const float bv_ = bias[gn];
      const int   h = gn >> 6, d = gn & 63;
#pragma unroll
      for (int r = 0; r < 8; ++r) {
        const int gm = m0 + wave * 32 + mt * 16 + (lane >> 4) * 8 + r;
        const int b  = gm >> 11;           // / SEQ
        const int s  = gm & (SEQ - 1);
        const float val = c[mt][nt][r] + bv_;
        if (z == 2) {
          vws[(((size_t)b * NH + h) * HD + d) * SEQ + s] = f2bf(val);
        } else {
          bf16* dst = (z == 0) ? qws : kws;
          dst[(((size_t)b * NH + h) * SEQ + s) * HD + d] = f2bf(val);
        }
      }
    }
  }
}

// ---------------------------------------------------------------------------
// Kernel 2: flash-style attention with TDM-staged, double-buffered K/V tiles.
// One wave owns 16 query rows; the block shares 32-key K/V tiles brought into
// LDS by the Tensor Data Mover (issued by wave 0, tracked on TENSORcnt) so
// the DMA of chunk j+1 overlaps the 8 WMMAs + softmax of chunk j.
// Grid: (SEQ/64, BATCH*NH), 128 threads (4 waves).
// ---------------------------------------------------------------------------
__global__ __launch_bounds__(128) void attn_kernel(
    const bf16* __restrict__ qws, const bf16* __restrict__ kws,
    const bf16* __restrict__ vws, const int* __restrict__ mask,
    float* __restrict__ out) {
  __shared__ __align__(16) bf16 Ks[2][32][64];   // keys: [s_local][d]
  __shared__ __align__(16) bf16 Vs[2][64][32];   // values: [d][s_local]
  __shared__ __align__(16) bf16 Ps[4][16][32];   // per-wave P staging

  const int lane  = threadIdx.x & 31;
  const int wave  = threadIdx.x >> 5;
  const int bh    = blockIdx.y;
  const int b     = bh / NH;
  const int h     = bh % NH;
  const int qbase = blockIdx.x * 64 + wave * 16;

  const bf16* Q = qws + (size_t)bh * SEQ * HD;   // [s][d]
  const bf16* K = kws + (size_t)bh * SEQ * HD;   // [s][d]
  const bf16* V = vws + (size_t)bh * HD * SEQ;   // [d][s]  (transposed)
  const int* msk = mask + (size_t)b * SEQ;

  // Q fragments for this wave's 16 rows (Dh=64 -> two K=32 A-frags)
  v16bf aQ0 = load_a32(Q + (size_t)qbase * HD,      HD, lane);
  v16bf aQ1 = load_a32(Q + (size_t)qbase * HD + 32, HD, lane);

  v8f acc[4];
#pragma unroll
  for (int g = 0; g < 4; ++g) acc[g] = zero8();
  float mrow[8], lrow[8];
#pragma unroll
  for (int r = 0; r < 8; ++r) { mrow[r] = -1e30f; lrow[r] = 0.f; }

  const float scale = 0.125f;   // 1/sqrt(64)
  const int   nch   = SEQ / 32;

#ifdef HAVE_TDM
  if (wave == 0) {   // prologue DMA: chunk 0 -> buffer 0
    tdm_load_2d((unsigned)(size_t)&Ks[0][0][0], K,          HD, SEQ, HD, 32, HD);
    tdm_load_2d((unsigned)(size_t)&Vs[0][0][0], V,          SEQ, HD, 32, HD, SEQ);
  }
#endif

  for (int j = 0; j < nch; ++j) {
    const int cur = j & 1;
    const int t0  = j * 32;

#ifdef HAVE_TDM
    if (wave == 0) __builtin_amdgcn_s_wait_tensorcnt((short)0);
    __syncthreads();                       // chunk j visible to all waves
    if (wave == 0 && j + 1 < nch) {        // overlap: DMA chunk j+1 into other buffer
      const int nxt = t0 + 32;
      tdm_load_2d((unsigned)(size_t)&Ks[cur ^ 1][0][0], K + (size_t)nxt * HD, HD, SEQ, HD, 32, HD);
      tdm_load_2d((unsigned)(size_t)&Vs[cur ^ 1][0][0], V + (size_t)nxt,      SEQ, HD, 32, HD, SEQ);
    }
#else
    // fallback: cooperative synchronous staging
    {
      const int row = threadIdx.x & 31, half = threadIdx.x >> 5;   // 4 halves x 16B
      *(v8bf*)&Ks[cur][row][half * 8] = *(const v8bf*)(K + (size_t)(t0 + row) * HD + half * 8);
      const int dr = threadIdx.x >> 1, q4 = (threadIdx.x & 1) * 16;
      *(v8bf*)&Vs[cur][dr][q4]     = *(const v8bf*)(V + (size_t)dr * SEQ + t0 + q4);
      *(v8bf*)&Vs[cur][dr][q4 + 8] = *(const v8bf*)(V + (size_t)dr * SEQ + t0 + q4 + 8);
      __syncthreads();
    }
#endif

    // ---- scores: 16 queries x 32 keys  (4 WMMAs) from LDS tiles ----
    v16bf b00 = load_b32(&Ks[cur][0][0],   HD, lane);
    v16bf b01 = load_b32(&Ks[cur][0][32],  HD, lane);
    v16bf b10 = load_b32(&Ks[cur][16][0],  HD, lane);
    v16bf b11 = load_b32(&Ks[cur][16][32], HD, lane);
    v8f s0 = zero8(), s1 = zero8();
    s0 = WMMA_BF16(aQ0, b00, s0);
    s0 = WMMA_BF16(aQ1, b01, s0);
    s1 = WMMA_BF16(aQ0, b10, s1);
    s1 = WMMA_BF16(aQ1, b11, s1);

    // additive mask (column = key index = lane&15 within each 16-key frag)
    const float add0 = (1.0f - (float)msk[t0 + (lane & 15)])      * -10000.f;
    const float add1 = (1.0f - (float)msk[t0 + 16 + (lane & 15)]) * -10000.f;

    // ---- online softmax (rows in VGPR index; cols across 16-lane half) ----
    float p0[8], p1[8], alpha[8];
#pragma unroll
    for (int r = 0; r < 8; ++r) {
      float x0 = s0[r] * scale + add0;
      float x1 = s1[r] * scale + add1;
      float mx = fmaxf(x0, x1);
#pragma unroll
      for (int sh = 1; sh < 16; sh <<= 1)
        mx = fmaxf(mx, __shfl_xor(mx, sh, 32));
      float mnew = fmaxf(mrow[r], mx);
      alpha[r] = __expf(mrow[r] - mnew);
      mrow[r]  = mnew;
      p0[r] = __expf(x0 - mnew);
      p1[r] = __expf(x1 - mnew);
      float ps = p0[r] + p1[r];
#pragma unroll
      for (int sh = 1; sh < 16; sh <<= 1)
        ps += __shfl_xor(ps, sh, 32);
      lrow[r] = lrow[r] * alpha[r] + ps;
    }
#pragma unroll
    for (int g = 0; g < 4; ++g)
#pragma unroll
      for (int r = 0; r < 8; ++r) acc[g][r] *= alpha[r];

    // ---- P: C-layout -> LDS -> A-layout ----
#pragma unroll
    for (int r = 0; r < 8; ++r) {
      Ps[wave][(lane >> 4) * 8 + r][lane & 15]        = f2bf(p0[r]);
      Ps[wave][(lane >> 4) * 8 + r][(lane & 15) + 16] = f2bf(p1[r]);
    }
    asm volatile("s_wait_dscnt 0" ::: "memory");   // CDNA5 split-counter wait
    v16bf aP = load_a32(&Ps[wave][0][0], 32, lane);

    // ---- P (16x32) @ V (32x64): 4 WMMAs ----
#pragma unroll
    for (int g = 0; g < 4; ++g) {
      v16bf bV = load_b32(&Vs[cur][g * 16][0], 32, lane);
      acc[g] = WMMA_BF16(aP, bV, acc[g]);
    }
    __syncthreads();   // all waves done with buffer `cur` before it is re-filled
  }

  // ---- epilogue: normalize, write f32 [b, s, h*64 + d] ----
#pragma unroll
  for (int r = 0; r < 8; ++r) {
    const int   s   = qbase + (lane >> 4) * 8 + r;
    const float inv = 1.0f / lrow[r];
    float* dst = out + ((size_t)b * SEQ + s) * DMODEL + h * HD;
#pragma unroll
    for (int g = 0; g < 4; ++g)
      dst[g * 16 + (lane & 15)] = acc[g][r] * inv;
  }
}

// ---------------------------------------------------------------------------
extern "C" void kernel_launch(void* const* d_in, const int* in_sizes, int n_in,
                              void* d_out, int out_size, void* d_ws, size_t ws_size,
                              hipStream_t stream) {
  const float* x  = (const float*)d_in[0];
  const float* Wq = (const float*)d_in[1];
  const float* bq = (const float*)d_in[2];
  const float* Wk = (const float*)d_in[3];
  const float* bk = (const float*)d_in[4];
  const float* Wv = (const float*)d_in[5];
  const float* bv = (const float*)d_in[6];
  const int*  msk = (const int*)d_in[7];
  float* out = (float*)d_out;

  const size_t QKV = (size_t)BATCH * SEQ * DMODEL;   // 6.29M elems (bf16)
  bf16* qws = (bf16*)d_ws;
  bf16* kws = qws + QKV;
  bf16* vws = kws + QKV;

  qkv_proj_kernel<<<dim3((BATCH * SEQ) / 128, DMODEL / 64, 3), 128, 0, stream>>>(
      x, Wq, bq, Wk, bk, Wv, bv, qws, kws, vws);
  attn_kernel<<<dim3(SEQ / 64, BATCH * NH), 128, 0, stream>>>(
      qws, kws, vws, msk, out);
}
